// RealAMSoftmax_77790447665631
// MI455X (gfx1250) — matrix operand order repaired and checked by reading, not compiled, not run on hardware
//
#include <hip/hip_runtime.h>

// ---------------------------------------------------------------------------
// AM-Softmax loss for MI455X (gfx1250, wave32, WMMA).
//   cos  = norm(E)[1024,512] @ norm(W)^T[512,100000]      (bf16 WMMA, f32 acc)
//   loss = mean_n log1p( sum_{c != t_n} relu(30*clip(cos) + 9 - phi_n) )
// Main GEMM uses v_wmma_f32_16x16x32_bf16. The bf16 normalized weight copy
// (102.4 MB) is L2-resident (192 MB L2), so after the first M-tile pass the
// GEMM runs on the WMMA roofline, not HBM (23.3 TB/s would only be ~9 us).
// Each wave processes TWO class tiles with two independent accumulator
// chains to keep the XDL pipe busy and double outstanding B loads.
// ---------------------------------------------------------------------------

typedef __attribute__((ext_vector_type(16))) __bf16 v16bf;
typedef __attribute__((ext_vector_type(8)))  float  v8f;

#define NUM_CLASSES 100000
#define EMBED       512
#define BATCH       1024
#define SCALEF      30.0f
#define MARG        9.0f            // SCALE * MARGIN
#define CLIP_LO     (-1.0f + 1e-7f)
#define CLIP_HI     ( 1.0f - 1e-7f)
#define CPAIRS      (NUM_CLASSES / 32)   // 3125 pairs of 16-class tiles
#define MTILES      (BATCH / 16)         // 64
#define GEMM_GX     40                   // class-pair groups

union FragBF {
    uint4  u[2];
    v16bf  v;
};

__device__ __forceinline__ unsigned short f2bf(float f) {
    union { float f; unsigned u; } v; v.f = f;
    unsigned u = v.u;
    u += 0x7FFFu + ((u >> 16) & 1u);      // round-to-nearest-even
    return (unsigned short)(u >> 16);
}
__device__ __forceinline__ float bf2f(unsigned short h) {
    union { unsigned u; float f; } v; v.u = ((unsigned)h) << 16;
    return v.f;
}
__device__ __forceinline__ float waveReduceAdd(float s) {
    s += __shfl_xor(s, 1);
    s += __shfl_xor(s, 2);
    s += __shfl_xor(s, 4);
    s += __shfl_xor(s, 8);
    s += __shfl_xor(s, 16);
    return s;
}

// --- 1. normalize weight rows -> bf16 (one wave32 per row) ------------------
__global__ __launch_bounds__(256) void norm_w_kernel(
    const float* __restrict__ w, unsigned short* __restrict__ wbf)
{
    const int row  = blockIdx.x * 8 + (threadIdx.x >> 5);   // 12500*8 = 100000
    const int lane = threadIdx.x & 31;
    const float* wr = w + (size_t)row * EMBED;
    float vals[16];
    float s = 0.0f;
#pragma unroll
    for (int i = 0; i < 16; ++i) {
        float v = wr[lane + 32 * i];
        vals[i] = v;
        s += v * v;
    }
    s = waveReduceAdd(s);
    const float inv = 1.0f / fmaxf(sqrtf(s), 1e-12f);
    unsigned short* o = wbf + (size_t)row * EMBED;
#pragma unroll
    for (int i = 0; i < 16; ++i)
        o[lane + 32 * i] = f2bf(vals[i] * inv);
}

// --- 2. normalize embedding rows -> bf16; zero row_sum ----------------------
__global__ __launch_bounds__(256) void norm_e_kernel(
    const float* __restrict__ e, unsigned short* __restrict__ ebf,
    float* __restrict__ row_sum)
{
    const int row  = blockIdx.x * 8 + (threadIdx.x >> 5);   // 128*8 = 1024
    const int lane = threadIdx.x & 31;
    const float* er = e + (size_t)row * EMBED;
    float vals[16];
    float s = 0.0f;
#pragma unroll
    for (int i = 0; i < 16; ++i) {
        float v = er[lane + 32 * i];
        vals[i] = v;
        s += v * v;
    }
    s = waveReduceAdd(s);
    const float inv = 1.0f / fmaxf(sqrtf(s), 1e-12f);
    unsigned short* o = ebf + (size_t)row * EMBED;
#pragma unroll
    for (int i = 0; i < 16; ++i)
        o[lane + 32 * i] = f2bf(vals[i] * inv);
    if (lane == 0) row_sum[row] = 0.0f;   // re-zeroed every call (graph-safe)
}

// --- 3. phi[n] = 30*clip(dot(e_n, w_{t_n})) with GEMM-matching bf16 math ----
__global__ __launch_bounds__(256) void phi_kernel(
    const unsigned short* __restrict__ ebf,
    const unsigned short* __restrict__ wbf,
    const int* __restrict__ gt, float* __restrict__ phi)
{
    const int row  = blockIdx.x * 8 + (threadIdx.x >> 5);   // 128*8 = 1024
    const int lane = threadIdx.x & 31;
    const int t    = gt[row];
    const unsigned short* er = ebf + (size_t)row * EMBED;
    const unsigned short* wr = wbf + (size_t)t   * EMBED;
    float s = 0.0f;
#pragma unroll
    for (int i = 0; i < 16; ++i) {
        const int idx = lane + 32 * i;
        s += bf2f(er[idx]) * bf2f(wr[idx]);
    }
    s = waveReduceAdd(s);
    if (lane == 0)
        phi[row] = fminf(fmaxf(s, CLIP_LO), CLIP_HI) * SCALEF;
}

// --- 4. main GEMM + masked-relu-margin row reduction ------------------------
__global__ __launch_bounds__(256) void amsoftmax_gemm_kernel(
    const unsigned short* __restrict__ wbf,   // [100000,512] bf16, L2-resident
    const unsigned short* __restrict__ ebf,   // [1024,512]  bf16
    const float* __restrict__ phi,            // [1024]
    const int* __restrict__ gt,               // [1024]
    float* __restrict__ row_sum)              // [1024]
{
    __shared__ uint4 ldsA[1024];              // 16 rows x 512 bf16 = 16 KB

    const int mtile = blockIdx.y;             // 0..63
    // Stage this block's A tile (16 batch rows x K=512) into LDS once.
    const uint4* asrc = (const uint4*)(ebf + (size_t)mtile * 16 * EMBED);
    for (int j = threadIdx.x; j < 1024; j += 256) ldsA[j] = asrc[j];
    __syncthreads();

    const int lane = threadIdx.x & 31;
    const int wave = threadIdx.x >> 5;
    const int half = lane >> 4;               // 0/1: which 16-lane group
    const int col  = lane & 15;               // N within tile
    const int rowbase = mtile * 16;

    // Hoist all 16 A fragments (K=512) into registers: 128 VGPRs, reused for
    // every class tile this wave touches. Layout per ISA (16-bit A 16x32):
    //   lanes 0-15  row=lane,    K = {base+0..7, base+16..23}
    //   lanes 16-31 row=lane-16, K = {base+8..15, base+24..31}
    FragBF afrag[16];
#pragma unroll
    for (int kk = 0; kk < 16; ++kk) {
        const int aoff = col * EMBED + kk * 32 + half * 8;  // bf16 elements
        afrag[kk].u[0] = ldsA[aoff >> 3];
        afrag[kk].u[1] = ldsA[(aoff + 16) >> 3];
    }

    // Per-row constants for the 8 C/D slots this lane owns (M = r + 8*half).
    float phi_r[8];
    int   gt_r[8];
#pragma unroll
    for (int r = 0; r < 8; ++r) {
        const int row = rowbase + r + 8 * half;
        phi_r[r] = phi[row];
        gt_r[r]  = gt[row];
    }
    float rsum[8] = {0.0f, 0.0f, 0.0f, 0.0f, 0.0f, 0.0f, 0.0f, 0.0f};

    // Two class tiles (32 classes) per iteration: two independent WMMA
    // accumulation chains + 4 outstanding b128 B-loads per K-step.
    for (int cp = blockIdx.x * 8 + wave; cp < CPAIRS; cp += GEMM_GX * 8) {
        v8f acc0 = {0.0f, 0.0f, 0.0f, 0.0f, 0.0f, 0.0f, 0.0f, 0.0f};
        v8f acc1 = {0.0f, 0.0f, 0.0f, 0.0f, 0.0f, 0.0f, 0.0f, 0.0f};
        const unsigned short* wrow0 = wbf + (size_t)cp * 32 * EMBED;
        const unsigned short* wrow1 = wrow0 + (size_t)16 * EMBED;
#pragma unroll
        for (int kk = 0; kk < 16; ++kk) {     // K = 512 in steps of 32
            // B fragment (32x16 bf16): lane = class column, K = {half*8+0..7,
            // 16+half*8+0..7} (matches A) -> two b128 loads.
            const size_t boff = (size_t)col * EMBED + kk * 32 + half * 8;
            FragBF b0, b1;
            const uint4* bp0 = (const uint4*)(wrow0 + boff);
            const uint4* bp1 = (const uint4*)(wrow1 + boff);
            b0.u[0] = bp0[0]; b0.u[1] = bp0[2];
            b1.u[0] = bp1[0]; b1.u[1] = bp1[2];
            acc0 = __builtin_amdgcn_wmma_f32_16x16x32_bf16(
                false, afrag[kk].v, false, b0.v, (short)0, acc0, false, false);
            acc1 = __builtin_amdgcn_wmma_f32_16x16x32_bf16(
                false, afrag[kk].v, false, b1.v, (short)0, acc1, false, false);
        }
        // Epilogue: clip*scale + margin - phi, mask target column, relu.
        const int cclass0 = cp * 32 + col;
        const int cclass1 = cclass0 + 16;
#pragma unroll
        for (int r = 0; r < 8; ++r) {
            float c0 = fminf(fmaxf(acc0[r], CLIP_LO), CLIP_HI) * SCALEF;
            float x0 = c0 + MARG - phi_r[r];
            rsum[r] += ((cclass0 != gt_r[r]) && (x0 > 0.0f)) ? x0 : 0.0f;
            float c1 = fminf(fmaxf(acc1[r], CLIP_LO), CLIP_HI) * SCALEF;
            float x1 = c1 + MARG - phi_r[r];
            rsum[r] += ((cclass1 != gt_r[r]) && (x1 > 0.0f)) ? x1 : 0.0f;
        }
    }
    // Row sums: reduce across the 16 lanes of each half-group, then atomic.
#pragma unroll
    for (int r = 0; r < 8; ++r) {
        float v = rsum[r];
        v += __shfl_xor(v, 1);
        v += __shfl_xor(v, 2);
        v += __shfl_xor(v, 4);
        v += __shfl_xor(v, 8);
        if (col == 0)
            atomicAdd(&row_sum[rowbase + r + 8 * half], v);
    }
}

// --- 5. loss = mean(log1p(row_sum)) -----------------------------------------
__global__ __launch_bounds__(1024) void finalize_kernel(
    const float* __restrict__ row_sum, float* __restrict__ out)
{
    __shared__ float sm[32];
    const int t = threadIdx.x;
    float v = log1pf(row_sum[t]);
    v = waveReduceAdd(v);
    if ((t & 31) == 0) sm[t >> 5] = v;
    __syncthreads();
    if (t < 32) {
        float x = sm[t];
        x = waveReduceAdd(x);
        if (t == 0) out[0] = x * (1.0f / (float)BATCH);
    }
}

extern "C" void kernel_launch(void* const* d_in, const int* in_sizes, int n_in,
                              void* d_out, int out_size, void* d_ws, size_t ws_size,
                              hipStream_t stream)
{
    const float* embedding = (const float*)d_in[0];   // [1024,512]
    const int*   gtruth    = (const int*)  d_in[1];   // [1024]
    const float* weight    = (const float*)d_in[2];   // [100000,512]
    float*       out       = (float*)d_out;

    // Workspace layout (all 16B-aligned):
    //   wbf : 100000*512*2 = 102,400,000 B   (bf16 normalized weight)
    //   ebf : 1024*512*2   =   1,048,576 B   (bf16 normalized embedding)
    //   phi : 1024*4       =       4,096 B
    //   rsum: 1024*4       =       4,096 B
    char* ws = (char*)d_ws;
    unsigned short* wbf = (unsigned short*)(ws);
    unsigned short* ebf = (unsigned short*)(ws + 102400000);
    float* phi          = (float*)(ws + 102400000 + 1048576);
    float* row_sum      = (float*)(ws + 102400000 + 1048576 + 4096);

    norm_w_kernel<<<NUM_CLASSES / 8, 256, 0, stream>>>(weight, wbf);
    norm_e_kernel<<<BATCH / 8, 256, 0, stream>>>(embedding, ebf, row_sum);
    phi_kernel<<<BATCH / 8, 256, 0, stream>>>(ebf, wbf, gtruth, phi);
    amsoftmax_gemm_kernel<<<dim3(GEMM_GX, MTILES), 256, 0, stream>>>(
        wbf, ebf, phi, gtruth, row_sum);
    finalize_kernel<<<1, 1024, 0, stream>>>(row_sum, out);
}
